// QuantLinear_10101763080230
// MI455X (gfx1250) — matrix operand order repaired and checked by reading, not compiled, not run on hardware
//
#include <hip/hip_runtime.h>

typedef __attribute__((ext_vector_type(16))) _Float16 v16h;
typedef __attribute__((ext_vector_type(8)))  _Float16 v8h;
typedef __attribute__((ext_vector_type(4)))  _Float16 v4h;
typedef __attribute__((ext_vector_type(2)))  _Float16 v2h;
typedef __attribute__((ext_vector_type(8)))  float    v8f;
typedef __attribute__((ext_vector_type(4)))  float    v4f;

#define K_TOT 4096
#define N_TOT 4096
#define BM 128
#define BN 128
#define BK 64
#define LDA (BK + 8)   // halves: 144-byte row stride (16B aligned, bank-skewed)
#define LDB (BK + 8)   // halves: sB is n-major [BN][BK]

__global__ __launch_bounds__(256, 2)
void quantlinear_wmma(const float* __restrict__ X,
                      const int*   __restrict__ QW,
                      const float* __restrict__ scales,
                      const float* __restrict__ zeros,
                      const float* __restrict__ bias,
                      float* __restrict__ Y)
{
    __shared__ _Float16 sA[BM * LDA];   // 18 KB
    __shared__ _Float16 sB[BN * LDB];   // 18 KB

    const int tid  = threadIdx.x;
    const int lane = tid & 31;
    const int wave = tid >> 5;
    const int hi   = lane >> 4;       // K-half selector for frags
    const int lm   = lane & 15;

    const int m0 = blockIdx.y * BM;
    const int n0 = blockIdx.x * BN;

    // wave tiling: 4 waves along M, 2 along N; each wave owns 32(M) x 64(N)
    const int wm = (wave & 3) * 32;
    const int wn = (wave >> 2) * 64;

    // ---- B-fill constants: each thread owns one output channel of the tile
    const int bn  = tid & (BN - 1);     // 0..127
    const int brc = tid >> 7;           // 0..1
    const float scl  = scales[n0 + bn];
    const float zscl = __builtin_rintf(zeros[n0 + bn] / scl) * scl;

    // ---- A-fill constants
    const int ar  = tid >> 4;           // 0..15 row base
    const int ac4 = tid & 15;           // float4 slot within 64-wide K slab

    v8f acc[2][4];
    #pragma unroll
    for (int i = 0; i < 2; ++i)
        #pragma unroll
        for (int j = 0; j < 4; ++j)
            acc[i][j] = (v8f)0.f;

    for (int k0 = 0; k0 < K_TOT; k0 += BK) {
        // ---------- stage A tile: BM x BK fp32 -> f16 LDS (row-major)
        #pragma unroll
        for (int i = 0; i < 8; ++i) {
            const int row = ar + 16 * i;
            const v4f x4 = *reinterpret_cast<const v4f*>(
                X + (size_t)(m0 + row) * K_TOT + k0 + ac4 * 4);
            v4h h4;
            h4.x = (_Float16)x4.x; h4.y = (_Float16)x4.y;
            h4.z = (_Float16)x4.z; h4.w = (_Float16)x4.w;
            *reinterpret_cast<v4h*>(&sA[row * LDA + ac4 * 4]) = h4;
        }
        // ---------- stage B tile: dequant int4 -> f16 LDS (n-major)
        const int r0 = k0 >> 1;
        #pragma unroll
        for (int j = 0; j < 16; ++j) {
            const int rl = brc * 16 + j;                      // packed row 0..31
            const int q  = QW[(size_t)(r0 + rl) * N_TOT + n0 + bn];
            v2h w2;
            w2.x = (_Float16)((float)( q        & 0xF) * scl - zscl);
            w2.y = (_Float16)((float)((q >> 4)  & 0xF) * scl - zscl);
            *reinterpret_cast<v2h*>(&sB[bn * LDB + 2 * rl]) = w2;
        }
        __syncthreads();

        // hint next tiles toward the caches while the matrix units run
        if (k0 + BK < K_TOT) {
            __builtin_prefetch(X  + (size_t)(m0 + ar) * K_TOT + k0 + BK, 0, 0);
            __builtin_prefetch(QW + (size_t)((k0 + BK) >> 1) * N_TOT + n0 + bn, 0, 0);
        }

        // ---------- compute: two K=32 steps, 2x4 16x16 tiles per wave
        #pragma unroll
        for (int ks = 0; ks < BK; ks += 32) {
            v16h afrag[2], bfrag[4];
            #pragma unroll
            for (int i = 0; i < 2; ++i) {
                // A frag: row = lane%16; 16 contiguous halves at k=ks+hi*16
                const _Float16* ap = &sA[(wm + i * 16 + lm) * LDA + ks + hi * 16];
                v8h lo = *reinterpret_cast<const v8h*>(ap);
                v8h hh = *reinterpret_cast<const v8h*>(ap + 8);
                afrag[i] = __builtin_shufflevector(lo, hh,
                    0,1,2,3,4,5,6,7,8,9,10,11,12,13,14,15);
            }
            #pragma unroll
            for (int j = 0; j < 4; ++j) {
                // B frag: col = lane%16; 16 contiguous halves at k=ks+hi*16
                const _Float16* bp = &sB[(wn + j * 16 + lm) * LDB + ks + hi * 16];
                v8h lo = *reinterpret_cast<const v8h*>(bp);
                v8h hh = *reinterpret_cast<const v8h*>(bp + 8);
                bfrag[j] = __builtin_shufflevector(lo, hh,
                    0,1,2,3,4,5,6,7,8,9,10,11,12,13,14,15);
            }
            #pragma unroll
            for (int i = 0; i < 2; ++i)
                #pragma unroll
                for (int j = 0; j < 4; ++j)
                    acc[i][j] = __builtin_amdgcn_wmma_f32_16x16x32_f16(
                        false, afrag[i], false, bfrag[j],
                        (short)0, acc[i][j], false, false);
        }
        __syncthreads();
    }

    // ---------- epilogue: bias add + store (C/D layout: n=lane%16, m=hi*8+v)
    #pragma unroll
    for (int i = 0; i < 2; ++i) {
        const int mrow = m0 + wm + i * 16 + hi * 8;
        #pragma unroll
        for (int j = 0; j < 4; ++j) {
            const int ncol = n0 + wn + j * 16 + lm;
            const float bv = bias[ncol];
            #pragma unroll
            for (int v = 0; v < 8; ++v)
                Y[(size_t)(mrow + v) * N_TOT + ncol] = acc[i][j][v] + bv;
        }
    }
}

extern "C" void kernel_launch(void* const* d_in, const int* in_sizes, int n_in,
                              void* d_out, int out_size, void* d_ws, size_t ws_size,
                              hipStream_t stream) {
    (void)n_in; (void)out_size; (void)d_ws; (void)ws_size;
    const float* X      = (const float*)d_in[0];
    const int*   QW     = (const int*)  d_in[1];
    const float* scales = (const float*)d_in[2];
    const float* zeros  = (const float*)d_in[3];
    const float* bias   = (const float*)d_in[4];
    float* Y = (float*)d_out;

    const int M = in_sizes[0] / K_TOT;          // B*S = 8192
    dim3 grid(N_TOT / BN, M / BM);              // (32, 64)
    quantlinear_wmma<<<grid, 256, 0, stream>>>(X, QW, scales, zeros, bias, Y);
}
